// attention_59485297050243
// MI455X (gfx1250) — matrix operand order repaired and checked by reading, not compiled, not run on hardware
//
#include <hip/hip_runtime.h>
#include <hip/hip_bf16.h>

// ---------------------------------------------------------------------------
// Additive (Bahdanau) attention, fused flash-style single pass over `values`.
// B=64, T=2048, D=512, U=512.
// ---------------------------------------------------------------------------

typedef __attribute__((ext_vector_type(16))) __bf16 v16bf;
typedef __attribute__((ext_vector_type(8)))  float  v8f;

#define B_  64
#define T_  2048
#define D_  512
#define U_  512
#define SEGS 4
#define TSEG (T_ / SEGS)   // 512
#define TILE 64            // T rows per LDS tile (4 WMMA M-tiles)

union Frag32B {
    uint4 u4[2];
    v16bf v;
};

__device__ __forceinline__ unsigned f2bf_pack(float a, float b) {
    unsigned ua = __float_as_uint(a);
    unsigned ub = __float_as_uint(b);
    ua = (ua + 0x7FFFu + ((ua >> 16) & 1u)) >> 16;   // RNE truncate to bf16
    ub = (ub + 0x7FFFu + ((ub >> 16) & 1u)) >> 16;
    return (ua & 0xFFFFu) | (ub << 16);
}

__device__ __forceinline__ float bf2f(unsigned hw) {
    return __uint_as_float(hw << 16);
}

__device__ __forceinline__ float fast_tanh(float x) {
    float cx = fminf(15.f, fmaxf(-15.f, x));
    float e  = __expf(2.f * cx);          // v_exp_f32
    return (e - 1.f) / (e + 1.f);
}

// ---------------------------------------------------------------------------
// Kernel 0: W2 fp32 -> bf16 (row-major [D,U]); stays resident in L2 (512 KB).
// ---------------------------------------------------------------------------
__global__ void convW2_kernel(const float* __restrict__ W2,
                              unsigned short* __restrict__ W2bf) {
    int i = (blockIdx.x * 256 + threadIdx.x) * 4;   // 256 blocks cover 262144
    float4 v = *(const float4*)(W2 + i);
    uint2 p;
    p.x = f2bf_pack(v.x, v.y);
    p.y = f2bf_pack(v.z, v.w);
    *(uint2*)(W2bf + i) = p;
}

// ---------------------------------------------------------------------------
// Kernel 1: qpb[b][u] = (query[b] @ W1)[u] + b1[u] + b2[u]   (tiny: 33 MFLOP)
// ---------------------------------------------------------------------------
__global__ void qproj_kernel(const float* __restrict__ query,
                             const float* __restrict__ W1,
                             const float* __restrict__ b1,
                             const float* __restrict__ b2,
                             float* __restrict__ qpb) {
    __shared__ float qs[D_];
    int b = blockIdx.x, tid = threadIdx.x;
    for (int d = tid; d < D_; d += 256) qs[d] = query[b * D_ + d];
    __syncthreads();
    for (int u = tid; u < U_; u += 256) {
        float acc = 0.f;
        #pragma unroll 8
        for (int d = 0; d < D_; ++d)
            acc = fmaf(qs[d], W1[d * U_ + u], acc);
        qpb[b * U_ + u] = acc + b1[u] + b2[u];
    }
}

// ---------------------------------------------------------------------------
// Kernel 2: main fused kernel. One workgroup = (batch b, segment seg).
//   For each 64-row tile: stage values->bf16 in LDS (+mask), WMMA GEMM vs W2bf
//   in TWO N-passes (32 cols each) to keep VGPR pressure low, tanh + Wv
//   reduction -> scores, online-softmax rescale, context accumulate from the
//   SAME LDS tile. Single HBM pass over `values`.
// ---------------------------------------------------------------------------
__global__ void __launch_bounds__(256)
attn_main_kernel(const float* __restrict__ values,
                 const unsigned short* __restrict__ W2bf,
                 const float* __restrict__ qpb,
                 const float* __restrict__ Wv,
                 const float* __restrict__ bv,
                 float* __restrict__ scoresBuf,   // [B,T] masked raw scores
                 float* __restrict__ ctxPart,     // [B,SEGS,D]
                 float* __restrict__ segStats) {  // [B,SEGS,2] = (m, s)
    extern __shared__ unsigned char smem[];
    unsigned short* Atile = (unsigned short*)smem;          // 64 x 512 bf16 = 64 KB
    float*    scoresS = (float*)(smem + TILE * D_ * 2);     // [64]
    float*    pbufS   = scoresS + TILE;                     // [64]
    unsigned* flagsS  = (unsigned*)(pbufS + TILE);          // [64]
    float*    statsS  = (float*)(flagsS + TILE);            // [0]=m [1]=s [2]=alpha [3]=newM

    const int b    = blockIdx.x >> 2;
    const int seg  = blockIdx.x & 3;
    const int tid  = threadIdx.x;
    const int wave = tid >> 5;
    const int lane = tid & 31;
    const int l15  = lane & 15;
    const int khalf = lane >> 4;

    const float bvv = bv[0];

    // Per-wave epilogue constants: lane's N column within each of 4 N-tiles
    float qv[4], wvv[4];
    #pragma unroll
    for (int j = 0; j < 4; ++j) {
        int n = wave * 64 + j * 16 + l15;
        qv[j]  = qpb[b * U_ + n];
        wvv[j] = Wv[n];
    }

    if (tid == 0) { statsS[0] = -3.0e38f; statsS[1] = 0.f; }

    float2 cacc = make_float2(0.f, 0.f);   // context accumulator: d = 2*tid, 2*tid+1
    const int d0 = tid * 2;

    const v8f zero8 = {0.f, 0.f, 0.f, 0.f, 0.f, 0.f, 0.f, 0.f};

    for (int tile = 0; tile < TSEG / TILE; ++tile) {
        const int t0 = seg * TSEG + tile * TILE;

        __syncthreads();   // previous tile fully consumed
        if (tid < TILE) { flagsS[tid] = 0u; scoresS[tid] = 0.f; }
        __syncthreads();

        // ---- Stage fp32 values -> bf16 LDS tile; compute per-row nonzero mask
        {
            const int row = tid >> 2;
            const int cb  = (tid & 3) * 128;
            const float4* src = (const float4*)(values +
                (((size_t)b * T_ + t0 + row) * D_ + cb));
            uint2* dst = (uint2*)(Atile + row * D_ + cb);
            bool nz = false;
            #pragma unroll 8
            for (int i = 0; i < 32; ++i) {
                float4 v = src[i];
                nz |= (v.x != 0.f) | (v.y != 0.f) | (v.z != 0.f) | (v.w != 0.f);
                uint2 p;
                p.x = f2bf_pack(v.x, v.y);
                p.y = f2bf_pack(v.z, v.w);
                dst[i] = p;
            }
            if (nz) atomicOr(&flagsS[row], 1u);
        }
        __syncthreads();

        // ---- GEMM in two N-passes (2 x 16 columns each) to limit VGPR use
        #pragma unroll 1
        for (int jp = 0; jp < 2; ++jp) {
            v8f acc[4][2];
            #pragma unroll
            for (int i = 0; i < 4; ++i)
                #pragma unroll
                for (int j = 0; j < 2; ++j) acc[i][j] = zero8;

            #pragma unroll 2
            for (int k0 = 0; k0 < D_; k0 += 32) {
                // A frag layout (16-bit A 16x32): lanes 0-15 row M; VGPR0-3 =
                // K k0..k0+7 (lanes 0-15) / k0+8..15 (lanes 16-31); VGPR4-7 = +16.
                Frag32B a[4];
                #pragma unroll
                for (int i = 0; i < 4; ++i) {
                    const unsigned short* ap =
                        Atile + (i * 16 + l15) * D_ + k0 + khalf * 8;
                    a[i].u4[0] = *(const uint4*)ap;
                    a[i].u4[1] = *(const uint4*)(ap + 16);
                }
                const int krow = k0 + khalf * 8;   // B: lane -> N column, elements over K (table order)
                #pragma unroll
                for (int j = 0; j < 2; ++j) {
                    const unsigned short* bp =
                        W2bf + (size_t)krow * U_ + wave * 64 + (jp * 2 + j) * 16 + l15;
                    Frag32B bf;
                    unsigned bw[8];
                    #pragma unroll
                    for (int q = 0; q < 8; ++q) {
                        const int kk = (q & 3) * 2 + (q >> 2) * 16;
                        bw[q] = (unsigned)bp[(size_t)kk * U_] | ((unsigned)bp[(size_t)(kk + 1) * U_] << 16);
                    }
                    bf.u4[0] = make_uint4(bw[0], bw[1], bw[2], bw[3]);  // K = 8h+0..7
                    bf.u4[1] = make_uint4(bw[4], bw[5], bw[6], bw[7]);  // K = 16+8h+0..7
                    #pragma unroll
                    for (int i = 0; i < 4; ++i) {
                        acc[i][j] = __builtin_amdgcn_wmma_f32_16x16x32_bf16(
                            false, a[i].v, false, bf.v, (short)0, acc[i][j],
                            false, false);
                    }
                }
            }

            // ---- Pass epilogue: sum_n tanh(vproj + qpb[n]) * Wv[n] -> LDS
            float part[4][8];
            #pragma unroll
            for (int i = 0; i < 4; ++i)
                #pragma unroll
                for (int r = 0; r < 8; ++r) part[i][r] = 0.f;

            #pragma unroll
            for (int j = 0; j < 2; ++j) {
                const float qvj = qv[jp * 2 + j];
                const float wvj = wvv[jp * 2 + j];
                #pragma unroll
                for (int i = 0; i < 4; ++i) {
                    #pragma unroll
                    for (int r = 0; r < 8; ++r) {
                        float h = fast_tanh(acc[i][j][r] + qvj);
                        part[i][r] = fmaf(h, wvj, part[i][r]);
                    }
                }
            }
            // reduce over the 16 lanes of each half (N dimension)
            #pragma unroll
            for (int i = 0; i < 4; ++i) {
                #pragma unroll
                for (int r = 0; r < 8; ++r) {
                    float v = part[i][r];
                    v += __shfl_xor(v, 1);
                    v += __shfl_xor(v, 2);
                    v += __shfl_xor(v, 4);
                    v += __shfl_xor(v, 8);
                    if (l15 == 0) {
                        int row = i * 16 + r + (khalf << 3);
                        atomicAdd(&scoresS[row], v);   // ds_add_f32, 8 waves x 2
                    }
                }
            }
        }
        __syncthreads();

        // ---- Masked scores -> global (for final weight pass)
        if (tid < TILE) {
            bool ok = flagsS[tid] != 0u;
            float sc = ok ? (scoresS[tid] + bvv) : -3.0e38f;
            scoresS[tid] = sc;
            scoresBuf[(size_t)b * T_ + t0 + tid] = sc;
        }
        __syncthreads();

        // ---- Online softmax: new running max + rescale factor (wave 0)
        if (wave == 0) {
            float v = fmaxf(scoresS[lane], scoresS[lane + 32]);
            v = fmaxf(v, __shfl_xor(v, 16));
            v = fmaxf(v, __shfl_xor(v, 8));
            v = fmaxf(v, __shfl_xor(v, 4));
            v = fmaxf(v, __shfl_xor(v, 2));
            v = fmaxf(v, __shfl_xor(v, 1));
            if (lane == 0) {
                float m0 = statsS[0];
                float tm = fmaxf(m0, v);
                statsS[2] = __expf(m0 - tm);   // alpha
                statsS[3] = tm;                // newM
                statsS[0] = tm;
            }
        }
        __syncthreads();
        const float alpha = statsS[2];
        const float newM  = statsS[3];
        if (tid < TILE) {
            float p = (flagsS[tid] != 0u) ? __expf(scoresS[tid] - newM) : 0.f;
            pbufS[tid] = p;
        }
        cacc.x *= alpha;
        cacc.y *= alpha;
        __syncthreads();
        // running denominator update (wave 0); others proceed to context accum
        if (wave == 0) {
            float v = pbufS[lane] + pbufS[lane + 32];
            v += __shfl_xor(v, 16);
            v += __shfl_xor(v, 8);
            v += __shfl_xor(v, 4);
            v += __shfl_xor(v, 2);
            v += __shfl_xor(v, 1);
            if (lane == 0) statsS[1] = statsS[1] * alpha + v;
        }

        // ---- Context accumulate straight from the LDS bf16 tile
        {
            const unsigned short* col = Atile + d0;
            #pragma unroll 8
            for (int r = 0; r < TILE; ++r) {
                float p = pbufS[r];
                unsigned w = *(const unsigned*)(col + r * D_);
                cacc.x = fmaf(p, bf2f(w & 0xFFFFu), cacc.x);
                cacc.y = fmaf(p, bf2f(w >> 16), cacc.y);
            }
        }
    }

    // ---- Segment partial outputs
    {
        size_t cbase = (((size_t)b * SEGS + seg) * D_) + d0;
        ctxPart[cbase]     = cacc.x;
        ctxPart[cbase + 1] = cacc.y;
    }
    __syncthreads();
    if (tid == 0) {
        segStats[((size_t)b * SEGS + seg) * 2 + 0] = statsS[0];
        segStats[((size_t)b * SEGS + seg) * 2 + 1] = statsS[1];
    }
}

// ---------------------------------------------------------------------------
// Kernel 3: merge segment partials -> context [B,D] + weights [B,T]
// ---------------------------------------------------------------------------
__global__ void attn_finalize_kernel(const float* __restrict__ scoresBuf,
                                     const float* __restrict__ ctxPart,
                                     const float* __restrict__ segStats,
                                     float* __restrict__ out) {
    __shared__ float sFac[SEGS];
    __shared__ float sM, sInv;
    int b = blockIdx.x, tid = threadIdx.x;
    if (tid == 0) {
        float M = -3.0e38f;
        for (int s = 0; s < SEGS; ++s)
            M = fmaxf(M, segStats[((size_t)b * SEGS + s) * 2 + 0]);
        float tot = 0.f;
        for (int s = 0; s < SEGS; ++s) {
            float f = __expf(segStats[((size_t)b * SEGS + s) * 2 + 0] - M);
            sFac[s] = f;
            tot = fmaf(segStats[((size_t)b * SEGS + s) * 2 + 1], f, tot);
        }
        sM = M;
        sInv = 1.f / fmaxf(tot, 1e-30f);
    }
    __syncthreads();
    const float inv = sInv, M = sM;
    for (int d = tid; d < D_; d += 256) {
        float c = 0.f;
        #pragma unroll
        for (int s = 0; s < SEGS; ++s)
            c = fmaf(ctxPart[((size_t)b * SEGS + s) * D_ + d], sFac[s], c);
        out[(size_t)b * D_ + d] = c * inv;
    }
    for (int t = tid; t < T_; t += 256) {
        float sc = scoresBuf[(size_t)b * T_ + t];
        float w = (sc <= -1.0e37f) ? 0.f : __expf(sc - M) * inv;
        out[(size_t)B_ * D_ + (size_t)b * T_ + t] = w;
    }
}

// ---------------------------------------------------------------------------
extern "C" void kernel_launch(void* const* d_in, const int* in_sizes, int n_in,
                              void* d_out, int out_size, void* d_ws, size_t ws_size,
                              hipStream_t stream) {
    const float* query  = (const float*)d_in[0];
    const float* values = (const float*)d_in[1];
    const float* W1     = (const float*)d_in[2];
    const float* b1     = (const float*)d_in[3];
    const float* W2     = (const float*)d_in[4];
    const float* b2     = (const float*)d_in[5];
    const float* Wv     = (const float*)d_in[6];
    const float* bv     = (const float*)d_in[7];
    float* out = (float*)d_out;

    // Workspace layout
    char* w = (char*)d_ws;
    unsigned short* W2bf = (unsigned short*)w;                 // 512 KB
    float* qpb       = (float*)(w + 524288);                   // 128 KB
    float* scoresBuf = (float*)(w + 524288 + 131072);          // 512 KB
    float* ctxPart   = (float*)(w + 524288 + 131072 + 524288); // 512 KB
    float* segStats  = (float*)(w + 524288 + 131072 + 524288 + 524288); // 2 KB
    (void)in_sizes; (void)n_in; (void)out_size; (void)ws_size;

    convW2_kernel<<<256, 256, 0, stream>>>(W2, W2bf);
    qproj_kernel<<<B_, 256, 0, stream>>>(query, W1, b1, b2, qpb);

    const size_t smemBytes = (size_t)TILE * D_ * 2      // bf16 tile
                           + TILE * sizeof(float) * 2   // scores + pbuf
                           + TILE * sizeof(unsigned)    // flags
                           + 8 * sizeof(float);         // stats
    attn_main_kernel<<<B_ * SEGS, 256, smemBytes, stream>>>(
        values, W2bf, qpb, Wv, bv, scoresBuf, ctxPart, segStats);

    attn_finalize_kernel<<<B_, 256, 0, stream>>>(scoresBuf, ctxPart, segStats, out);
}
